// RNNModel_35313221108076
// MI455X (gfx1250) — matrix operand order repaired and checked by reading, not compiled, not run on hardware
//
#include <hip/hip_runtime.h>
#include <hip/hip_bf16.h>

// ---------------- types ----------------
typedef __attribute__((ext_vector_type(16))) __bf16 v16bf;
typedef __attribute__((ext_vector_type(8)))  __bf16 v8bf;
typedef __attribute__((ext_vector_type(8)))  float  v8f;
typedef __attribute__((ext_vector_type(4)))  int    v4i;
typedef __attribute__((address_space(1))) v4i* lm_gptr_b128;   // global <4 x i32>*
typedef __attribute__((address_space(3))) v4i* lm_lptr_b128;   // LDS    <4 x i32>*

// ---------------- problem dims ----------------
#define NTOK   33278
#define DINP   400
#define DHID   1150
#define TSEQ   70
#define BATCH  64
#define MTB    (TSEQ*BATCH)       // 4480, multiple of 16 (and of 128)
// padded dims (K -> mult of 32, N -> mult of 64)
#define KP_INP 416                // 400  -> 13*32
#define KP_HID 1152               // 1150 -> 36*32
#define NP_GHID 4608              // 4*1150=4600 -> 72*64
#define NP_GINP 1600              // 4*400        = 25*64
#define NP_VOC  33280             // 33278 -> 520*64

// LDS B-tile staging: 64 rows x 32 k (64B) padded to 80B row stride (bank-conflict-free)
#define LM_ROWB   80
#define LM_STAGEB (64 * LM_ROWB)  // 5120 B per stage

#ifndef __has_builtin
#define __has_builtin(x) 0
#endif
#if __has_builtin(__builtin_amdgcn_global_load_async_to_lds_b128) && \
    __has_builtin(__builtin_amdgcn_s_wait_asynccnt)
#define LM_ASYNC_LDS 1
#else
#define LM_ASYNC_LDS 0
#endif

// ============================================================
// f32 -> bf16 convert with zero padding: dst is Np x Kp, src Nv x Kv
// ============================================================
__global__ void lm_convert_pad(const float* __restrict__ src, __bf16* __restrict__ dst,
                               int Nv, int Kv, int Np, int Kp) {
    int i = blockIdx.x * 256 + threadIdx.x;
    int tot = Np * Kp;
    if (i >= tot) return;
    int n = i / Kp, k = i - n * Kp;
    float v = (n < Nv && k < Kv) ? src[(long)n * Kv + k] : 0.0f;
    dst[i] = (__bf16)v;
}

// ============================================================
// embedding gather -> bf16 padded activations (MTB x KP_INP)
// ============================================================
__global__ void lm_embed(const int* __restrict__ idx, const float* __restrict__ emb,
                         __bf16* __restrict__ x0) {
    int i = blockIdx.x * 256 + threadIdx.x;
    if (i >= MTB * KP_INP) return;
    int r = i / KP_INP, k = i - r * KP_INP;
    int tok = idx[r];
    float v = (k < DINP) ? emb[(long)tok * DINP + k] : 0.0f;
    x0[i] = (__bf16)v;
}

// ============================================================
// init h (bf16, zero-padded) and c (f32) from h0/c0 inputs
// ============================================================
__global__ void lm_init_state(const float* __restrict__ h0, const float* __restrict__ c0,
                              __bf16* __restrict__ h, float* __restrict__ c,
                              int n, int npad) {
    int i = blockIdx.x * 256 + threadIdx.x;
    if (i >= BATCH * npad) return;
    int b = i / npad, j = i - b * npad;
    float hv = (j < n) ? h0[b * n + j] : 0.0f;
    h[i] = (__bf16)hv;
    if (j < n) c[i] = c0[b * n + j];
}

// ============================================================
// bf16 WMMA GEMM with LDS-staged, double-buffered B operand.
//   C[M,N] = A[M,Kp] * B^T   (B stored N x Kp row-major = exactly WMMA-B layout)
// Block (256 thr = 8 waves) computes a 128M x 64N macro-tile:
//   - B 64x32 k-slab staged in LDS once per block via async global->LDS copy
//     (tracked by ASYNCcnt), consumed by all 8 waves via ds_load_b128
//     (80B row stride: conflict-free)
//   - each wave: 16 rows of A direct from global, 4 N-tiles, 4 WMMAs / k-step
// C: f32, row stride ldc, stores masked to col < Nvalid, optional bias[col].
// ============================================================
__global__ __launch_bounds__(256)
void lm_gemm_wmma(const __bf16* __restrict__ A, const __bf16* __restrict__ Bw,
                  float* __restrict__ C, const float* __restrict__ bias,
                  int M, int Np, int Kp, int Nvalid, int ldc) {
    __shared__ __align__(16) char smem[2 * LM_STAGEB];

    const int tid  = threadIdx.x;
    const int wid  = tid >> 5;
    const int lane = tid & 31;
    const int g    = lane >> 4;               // half-wave group (K sub-block)
    const int lm   = lane & 15;               // row (A) / column (B,C)

    const int nNblk = Np >> 6;
    const int mblk  = blockIdx.x / nNblk;
    const int nblk  = blockIdx.x - mblk * nNblk;
    const int n0    = nblk << 6;
    int mt = mblk * 8 + wid;                  // this wave's 16-row M tile
    const int mtMax = (M >> 4) - 1;
    if (mt > mtMax) mt = mtMax;               // duplicate work, never exit (barriers)
    const int m0 = mt << 4;

    // B copy role: thread moves one 16B chunk: row = tid>>2 (0..63), chunk = tid&3
    const int crow = tid >> 2, cch = tid & 3;
    const __bf16* gB = Bw + (long)(n0 + crow) * Kp + cch * 8;
    const int sOff = crow * LM_ROWB + cch * 16;

    const __bf16* Arow = A + (long)(m0 + lm) * Kp + g * 8;

    v8f acc0 = {}, acc1 = {}, acc2 = {}, acc3 = {};

#if LM_ASYNC_LDS
#define LM_COPY(KT)                                                              \
    __builtin_amdgcn_global_load_async_to_lds_b128(                              \
        (lm_gptr_b128)(v4i*)(gB + (KT) * 32),                                    \
        (lm_lptr_b128)(v4i*)(smem + ((KT)&1) * LM_STAGEB + sOff),                \
        0, 0);
#else
#define LM_COPY(KT)                                                              \
    { v8bf t_ = *(const v8bf*)(gB + (KT) * 32);                                  \
      *(v8bf*)(smem + ((KT)&1) * LM_STAGEB + sOff) = t_; }
#endif

    const int nk = Kp >> 5;
    LM_COPY(0)
    for (int kt = 0; kt < nk; ++kt) {
        if (kt + 1 < nk) {
            LM_COPY(kt + 1)
#if LM_ASYNC_LDS
            __builtin_amdgcn_s_wait_asynccnt(1);   // stage kt landed; kt+1 in flight
#endif
        } else {
#if LM_ASYNC_LDS
            __builtin_amdgcn_s_wait_asynccnt(0);
#endif
        }
        __syncthreads();                           // all waves' copies visible

        const char* bb = smem + (kt & 1) * LM_STAGEB;
        const int k0 = kt << 5;

        // A fragment: lane holds row m0+lm, k = g*8+0..7 and 16+g*8+0..7
        v8bf alo = *(const v8bf*)(Arow + k0);
        v8bf ahi = *(const v8bf*)(Arow + k0 + 16);
        v16bf a = __builtin_shufflevector(alo, ahi,
                    0,1,2,3,4,5,6,7, 8,9,10,11,12,13,14,15);

        // B fragments from LDS: tile j -> row 16j+lm, bytes [g*16, g*16+15] and [32+g*16, 32+g*16+15]
#define LM_BLOAD(BV, J)                                                          \
        v16bf BV;                                                                \
        {   const char* p_ = bb + (16*(J) + lm) * LM_ROWB + g * 16;              \
            v8bf bl_ = *(const v8bf*)(p_);                                       \
            v8bf bh_ = *(const v8bf*)(p_ + 32);                                  \
            BV = __builtin_shufflevector(bl_, bh_,                               \
                   0,1,2,3,4,5,6,7, 8,9,10,11,12,13,14,15); }
        LM_BLOAD(b0, 0)
        LM_BLOAD(b1, 1)
        LM_BLOAD(b2, 2)
        LM_BLOAD(b3, 3)
#undef LM_BLOAD

        acc0 = __builtin_amdgcn_wmma_f32_16x16x32_bf16(false, a, false, b0, (short)0, acc0, false, false);
        acc1 = __builtin_amdgcn_wmma_f32_16x16x32_bf16(false, a, false, b1, (short)0, acc1, false, false);
        acc2 = __builtin_amdgcn_wmma_f32_16x16x32_bf16(false, a, false, b2, (short)0, acc2, false, false);
        acc3 = __builtin_amdgcn_wmma_f32_16x16x32_bf16(false, a, false, b3, (short)0, acc3, false, false);

        __syncthreads();                           // protect buffer reuse
    }
#undef LM_COPY

    const int mrow = m0 + 8 * g;                   // VGPR r holds C[mrow+r][col]
#define LM_STORE_TILE(ACC, J)                                               \
    {                                                                       \
        int col = n0 + 16 * (J) + lm;                                       \
        if (col < Nvalid) {                                                 \
            float bv = bias ? bias[col] : 0.0f;                             \
            _Pragma("unroll")                                               \
            for (int r = 0; r < 8; ++r)                                     \
                C[(long)(mrow + r) * ldc + col] = (ACC)[r] + bv;            \
        }                                                                   \
    }
    LM_STORE_TILE(acc0, 0)
    LM_STORE_TILE(acc1, 1)
    LM_STORE_TILE(acc2, 2)
    LM_STORE_TILE(acc3, 3)
#undef LM_STORE_TILE
}

// ============================================================
// fused LSTM gate update for one time step
// pre = x_pre[t] + h_prev@Wh^T (hpre) + bi + bh
// f,i,o = sigmoid(pre[0:3n]); gg = tanh(pre[3n:4n])
// c = f*c + i*gg; h = o*tanh(c)  -> h stored bf16 (state + layer output)
// ============================================================
__global__ void lm_lstm_gates(const float* __restrict__ xpre_t,  // 64 rows, stride Np
                              const float* __restrict__ hpre,    // 64 rows, stride Np
                              const float* __restrict__ bi, const float* __restrict__ bh,
                              float* __restrict__ c,             // 64 x npad
                              __bf16* __restrict__ h,            // 64 x npad (state)
                              __bf16* __restrict__ xs_t,         // 64 x npad (sequence out)
                              int n, int npad, int Np) {
    int i = blockIdx.x * 256 + threadIdx.x;
    if (i >= BATCH * n) return;
    int b = i / n, j = i - b * n;
    long rb = (long)b * Np;
    float pf = xpre_t[rb + j]         + hpre[rb + j]         + bi[j]         + bh[j];
    float pi = xpre_t[rb + n + j]     + hpre[rb + n + j]     + bi[n + j]     + bh[n + j];
    float po = xpre_t[rb + 2*n + j]   + hpre[rb + 2*n + j]   + bi[2*n + j]   + bh[2*n + j];
    float pg = xpre_t[rb + 3*n + j]   + hpre[rb + 3*n + j]   + bi[3*n + j]   + bh[3*n + j];
    float fg = 1.0f / (1.0f + __expf(-pf));
    float ig = 1.0f / (1.0f + __expf(-pi));
    float og = 1.0f / (1.0f + __expf(-po));
    float gg = tanhf(pg);
    int ci = b * npad + j;
    float cn = fg * c[ci] + ig * gg;
    c[ci] = cn;
    float hn = og * tanhf(cn);
    __bf16 hb = (__bf16)hn;
    h[ci]    = hb;
    xs_t[ci] = hb;
}

// ============================================================
// host-side orchestration
// ============================================================
static inline int lm_blocks(long n) { return (int)((n + 255) / 256); }

extern "C" void kernel_launch(void* const* d_in, const int* in_sizes, int n_in,
                              void* d_out, int out_size, void* d_ws, size_t ws_size,
                              hipStream_t stream) {
    (void)in_sizes; (void)n_in; (void)out_size; (void)ws_size;

    const int*   inp   = (const int*)  d_in[0];
    const float* h0[3] = { (const float*)d_in[1], (const float*)d_in[3], (const float*)d_in[5] };
    const float* c0[3] = { (const float*)d_in[2], (const float*)d_in[4], (const float*)d_in[6] };
    const float* embw  = (const float*)d_in[7];
    const float* wi[3] = { (const float*)d_in[8],  (const float*)d_in[12], (const float*)d_in[16] };
    const float* bi[3] = { (const float*)d_in[9],  (const float*)d_in[13], (const float*)d_in[17] };
    const float* wh[3] = { (const float*)d_in[10], (const float*)d_in[14], (const float*)d_in[18] };
    const float* bh[3] = { (const float*)d_in[11], (const float*)d_in[15], (const float*)d_in[19] };
    const float* decb  = (const float*)d_in[20];
    float* out = (float*)d_out;

    // ---- workspace carve-up (256B aligned) ----
    size_t off = 0;
    char* base = (char*)d_ws;
    auto wsget = [&](size_t bytes) -> void* {
        off = (off + 255) & ~(size_t)255;
        void* p = base + off;
        off += bytes;
        return p;
    };
    __bf16* embB  = (__bf16*)wsget((size_t)NP_VOC  * KP_INP * 2);
    __bf16* wiB0  = (__bf16*)wsget((size_t)NP_GHID * KP_INP * 2);
    __bf16* whB0  = (__bf16*)wsget((size_t)NP_GHID * KP_HID * 2);
    __bf16* wiB1  = (__bf16*)wsget((size_t)NP_GHID * KP_HID * 2);
    __bf16* whB1  = (__bf16*)wsget((size_t)NP_GHID * KP_HID * 2);
    __bf16* wiB2  = (__bf16*)wsget((size_t)NP_GINP * KP_HID * 2);
    __bf16* whB2  = (__bf16*)wsget((size_t)NP_GINP * KP_INP * 2);
    __bf16* x0    = (__bf16*)wsget((size_t)MTB * KP_INP * 2);
    __bf16* xs1   = (__bf16*)wsget((size_t)MTB * KP_HID * 2);
    __bf16* xs2   = (__bf16*)wsget((size_t)MTB * KP_HID * 2);
    __bf16* xs3   = (__bf16*)wsget((size_t)MTB * KP_INP * 2);
    float*  xpre  = (float*) wsget((size_t)MTB * NP_GHID * 4);
    float*  hpre  = (float*) wsget((size_t)BATCH * NP_GHID * 4);
    __bf16* hcur  = (__bf16*)wsget((size_t)BATCH * KP_HID * 2);
    float*  ccur  = (float*) wsget((size_t)BATCH * KP_HID * 4);

    // ---- weight conversion to padded bf16 ----
    lm_convert_pad<<<lm_blocks((long)NP_VOC*KP_INP), 256, 0, stream>>>(embw,  embB, NTOK,   DINP, NP_VOC,  KP_INP);
    lm_convert_pad<<<lm_blocks((long)NP_GHID*KP_INP),256, 0, stream>>>(wi[0], wiB0, 4*DHID, DINP, NP_GHID, KP_INP);
    lm_convert_pad<<<lm_blocks((long)NP_GHID*KP_HID),256, 0, stream>>>(wh[0], whB0, 4*DHID, DHID, NP_GHID, KP_HID);
    lm_convert_pad<<<lm_blocks((long)NP_GHID*KP_HID),256, 0, stream>>>(wi[1], wiB1, 4*DHID, DHID, NP_GHID, KP_HID);
    lm_convert_pad<<<lm_blocks((long)NP_GHID*KP_HID),256, 0, stream>>>(wh[1], whB1, 4*DHID, DHID, NP_GHID, KP_HID);
    lm_convert_pad<<<lm_blocks((long)NP_GINP*KP_HID),256, 0, stream>>>(wi[2], wiB2, 4*DINP, DHID, NP_GINP, KP_HID);
    lm_convert_pad<<<lm_blocks((long)NP_GINP*KP_INP),256, 0, stream>>>(wh[2], whB2, 4*DINP, DINP, NP_GINP, KP_INP);

    // ---- embedding gather (writes padding zeros itself) ----
    lm_embed<<<lm_blocks((long)MTB*KP_INP), 256, 0, stream>>>(inp, embw, x0);

    // zero layer-output buffers so K-padding columns read as 0 in next GEMM
    (void)hipMemsetAsync(xs1, 0, (size_t)MTB * KP_HID * 2, stream);
    (void)hipMemsetAsync(xs2, 0, (size_t)MTB * KP_HID * 2, stream);
    (void)hipMemsetAsync(xs3, 0, (size_t)MTB * KP_INP * 2, stream);

    // ---- per-layer configuration ----
    const __bf16* Ain[3]  = { x0,   xs1,  xs2  };
    __bf16*       Aout[3] = { xs1,  xs2,  xs3  };
    const __bf16* WiB[3]  = { wiB0, wiB1, wiB2 };
    const __bf16* WhB[3]  = { whB0, whB1, whB2 };
    const int dinP[3]  = { KP_INP, KP_HID, KP_HID };
    const int dout[3]  = { DHID,   DHID,   DINP   };
    const int doutP[3] = { KP_HID, KP_HID, KP_INP };
    const int NpG[3]   = { NP_GHID, NP_GHID, NP_GINP };

    for (int l = 0; l < 3; ++l) {
        const int n = dout[l], npad = doutP[l], Np = NpG[l];

        lm_init_state<<<lm_blocks((long)BATCH*npad), 256, 0, stream>>>(
            h0[l], c0[l], hcur, ccur, n, npad);

        // time-batched input projection: xpre[MTB,Np] = xs @ Wi^T
        {
            int grid = ((MTB + 127) / 128) * (Np >> 6);
            lm_gemm_wmma<<<grid, 256, 0, stream>>>(
                Ain[l], WiB[l], xpre, nullptr, MTB, Np, dinP[l], Np, Np);
        }

        // sequential scan
        for (int t = 0; t < TSEQ; ++t) {
            int grid = ((BATCH + 127) / 128) * (Np >> 6);
            lm_gemm_wmma<<<grid, 256, 0, stream>>>(
                hcur, WhB[l], hpre, nullptr, BATCH, Np, npad, Np, Np);
            lm_lstm_gates<<<lm_blocks((long)BATCH*n), 256, 0, stream>>>(
                xpre + (size_t)t * BATCH * Np, hpre, bi[l], bh[l],
                ccur, hcur, Aout[l] + (size_t)t * BATCH * npad,
                n, npad, Np);
        }
    }

    // ---- decoder: out[MTB,NTOK] = xs3 @ emb^T + dec_b ----
    {
        int grid = ((MTB + 127) / 128) * (NP_VOC >> 6);
        lm_gemm_wmma<<<grid, 256, 0, stream>>>(
            xs3, embB, out, decb, MTB, NP_VOC, KP_INP, NTOK, NTOK);
    }
}
